// RoutingAttention_55138790146507
// MI455X (gfx1250) — matrix operand order repaired and checked by reading, not compiled, not run on hardware
//
#include <hip/hip_runtime.h>
#include <math.h>

typedef __attribute__((ext_vector_type(16))) _Float16 v16h;
typedef __attribute__((ext_vector_type(8)))  _Float16 v8h;
typedef __attribute__((ext_vector_type(8)))  float    v8f;

// Problem constants (from reference)
#define B_   2
#define P_   8
#define S_   1024        // K*Hp*W = 4*16*16
#define C_   256
#define H_   8
#define DK_  32
#define BHP_ (B_*H_*P_)  // 128
#define T_   (B_*P_*S_)  // 16384 tokens

__device__ inline v8f wmma_f16(v16h a, v16h b, v8f c) {
    // D = A(16x32 f16) * B(32x16 f16) + C(16x16 f32)
    return __builtin_amdgcn_wmma_f32_16x16x32_f16(
        /*neg_a=*/false, a, /*neg_b=*/false, b,
        /*c_mod=*/(short)0, c, /*reuse_a=*/false, /*reuse_b=*/false);
}

__device__ inline v8f zero8() {
    v8f z;
#pragma unroll
    for (int i = 0; i < 8; ++i) z[i] = 0.0f;
    return z;
}

// Load 2x8 consecutive f32 and convert to a v16h operand fragment.
__device__ inline v16h cvt_2x8(const float* p0, const float* p1) {
    float4 a = *reinterpret_cast<const float4*>(p0);
    float4 b = *reinterpret_cast<const float4*>(p0 + 4);
    float4 c = *reinterpret_cast<const float4*>(p1);
    float4 d = *reinterpret_cast<const float4*>(p1 + 4);
    v16h r;
    r[0]=(_Float16)a.x;  r[1]=(_Float16)a.y;  r[2]=(_Float16)a.z;  r[3]=(_Float16)a.w;
    r[4]=(_Float16)b.x;  r[5]=(_Float16)b.y;  r[6]=(_Float16)b.z;  r[7]=(_Float16)b.w;
    r[8]=(_Float16)c.x;  r[9]=(_Float16)c.y;  r[10]=(_Float16)c.z; r[11]=(_Float16)c.w;
    r[12]=(_Float16)d.x; r[13]=(_Float16)d.y; r[14]=(_Float16)d.z; r[15]=(_Float16)d.w;
    return r;
}

__device__ inline v16h combine(v8h lo, v8h hi) {
    v16h r;
#pragma unroll
    for (int i = 0; i < 8; ++i) { r[i] = lo[i]; r[8 + i] = hi[i]; }
    return r;
}

// ---------------------------------------------------------------------------
// Kernel 0: one-shot f32 -> f16 weight conversion (weights are reused by
// 1024 blocks each; convert once instead of per-tile).
// Grid: x covers 65536/ (256*8) , y = 4 selects {Wq,Wk,Wv,Wo}.
// Each thread converts 8 elements (two float4 loads -> 4 packed cvt).
// ---------------------------------------------------------------------------
__global__ __launch_bounds__(256) void cvt_weights(
    const float* __restrict__ Wq, const float* __restrict__ Wk,
    const float* __restrict__ Wv, const float* __restrict__ Wo,
    _Float16* __restrict__ Wqh, _Float16* __restrict__ Wkh,
    _Float16* __restrict__ Wvh, _Float16* __restrict__ Woh)
{
    const int m = blockIdx.y;
    const float* src = (m == 0) ? Wq : ((m == 1) ? Wk : ((m == 2) ? Wv : Wo));
    _Float16*    dst = (m == 0) ? Wqh : ((m == 1) ? Wkh : ((m == 2) ? Wvh : Woh));
    const int i = (blockIdx.x * 256 + threadIdx.x) * 8;   // element index
    float4 a = *reinterpret_cast<const float4*>(src + i);
    float4 b = *reinterpret_cast<const float4*>(src + i + 4);
    v8h r;
    r[0]=(_Float16)a.x; r[1]=(_Float16)a.y; r[2]=(_Float16)a.z; r[3]=(_Float16)a.w;
    r[4]=(_Float16)b.x; r[5]=(_Float16)b.y; r[6]=(_Float16)b.z; r[7]=(_Float16)b.w;
    *reinterpret_cast<v8h*>(dst + i) = r;
}

// ---------------------------------------------------------------------------
// Kernel 1: fused Q/K/V projection. y = x @ W^T + b, head-split on the fly.
//   qh/kh written as [b][h][p][s][dk] (f16), vT written transposed
//   [b][h][p][dk][s] (f16) so the PV B-operand loads are contiguous.
// Grid: x = T_/16 token tiles, y = 3 (q,k,v). Block = 128 (4 waves),
// each wave owns 4 output N-tiles of 16 dims. Weights already f16.
// ---------------------------------------------------------------------------
__global__ __launch_bounds__(128) void proj_qkv(
    const float* __restrict__ q, const float* __restrict__ k, const float* __restrict__ v,
    const _Float16* __restrict__ Wqh, const float* __restrict__ bq,
    const _Float16* __restrict__ Wkh, const float* __restrict__ bk,
    const _Float16* __restrict__ Wvh, const float* __restrict__ bv,
    _Float16* __restrict__ qh, _Float16* __restrict__ kh, _Float16* __restrict__ vT)
{
    const int mode = blockIdx.y;
    const float*    X    = (mode == 0) ? q   : ((mode == 1) ? k   : v);
    const _Float16* W    = (mode == 0) ? Wqh : ((mode == 1) ? Wkh : Wvh);
    const float*    bias = (mode == 0) ? bq  : ((mode == 1) ? bk  : bv);
    _Float16*       out  = (mode == 0) ? qh  : ((mode == 1) ? kh  : vT);

    const int tbase = blockIdx.x * 16;
    const int lane  = threadIdx.x & 31;
    const int wv    = threadIdx.x >> 5;
    const int nlo   = lane & 15;
    const int sel   = lane >> 4;

    v8f acc[4];
#pragma unroll
    for (int i = 0; i < 4; ++i) acc[i] = zero8();

    // A-operand: 16 tokens x 32 input channels per k-step (f32 -> f16 inline;
    // each activation row is read exactly once, so inline cvt is bandwidth-optimal).
    const float* xrow = X + (size_t)(tbase + nlo) * C_;
#pragma unroll
    for (int kk = 0; kk < 8; ++kk) {
        const int c0 = kk * 32 + sel * 16;                // A layout K-split
        v16h A = cvt_2x8(xrow + c0, xrow + c0 + 8);
#pragma unroll
        for (int nt = 0; nt < 4; ++nt) {
            const int obase = (wv * 4 + nt) * 16;
            // B-operand: B[K=c][N=o] = W[o][c]; lane holds column o. Direct f16.
            const _Float16* wrow = W + (size_t)(obase + nlo) * C_ + kk * 32 + sel * 16;
            v16h Bv = *reinterpret_cast<const v16h*>(wrow);
            acc[nt] = wmma_f16(A, Bv, acc[nt]);
        }
    }

    const int b   = tbase / (P_ * S_);
    const int p   = (tbase / S_) & (P_ - 1);
    const int s00 = tbase & (S_ - 1);
#pragma unroll
    for (int nt = 0; nt < 4; ++nt) {
        const int o  = (wv * 4 + nt) * 16 + nlo;
        const float bb = bias[o];
        const int h  = o >> 5;
        const int dk = o & 31;
        const size_t g = ((size_t)(b * H_ + h)) * P_ + p;
#pragma unroll
        for (int r = 0; r < 8; ++r) {
            const int srow = s00 + r + 8 * sel;
            const float val = acc[nt][r] + bb;
            size_t addr;
            if (mode == 2) addr = (g * DK_ + dk) * (size_t)S_ + srow;   // vT
            else           addr = (g * (size_t)S_ + srow) * DK_ + dk;   // qh/kh
            out[addr] = (_Float16)val;
        }
    }
}

// ---------------------------------------------------------------------------
// Kernel 2: flash-attention per (b,h,p) window. One wave per 16-query block.
// Grid: x = S_/16 = 64 row blocks, y = BHP_ = 128. Block = 32 threads.
// ---------------------------------------------------------------------------
__global__ __launch_bounds__(32) void attn(
    const _Float16* __restrict__ qh, const _Float16* __restrict__ kh,
    const _Float16* __restrict__ vT, _Float16* __restrict__ ctx)
{
    __shared__ __align__(32) _Float16 pbuf[16 * 32];

    const int lane = threadIdx.x & 31;
    const int nlo  = lane & 15;
    const int sel  = lane >> 4;
    const int bhp  = blockIdx.y;
    const int m0   = blockIdx.x * 16;

    const _Float16* Qb = qh + (size_t)bhp * S_ * DK_;
    const _Float16* Kb = kh + (size_t)bhp * S_ * DK_;
    const _Float16* Vb = vT + (size_t)bhp * DK_ * S_;

    // Q A-operand (16x32, dk is the K dim) — resident for the whole pass.
    const _Float16* qrow = Qb + (size_t)(m0 + nlo) * DK_ + sel * 16;
    const v16h Qa = combine(*reinterpret_cast<const v8h*>(qrow),
                            *reinterpret_cast<const v8h*>(qrow + 8));

    v8f O0 = zero8(), O1 = zero8();
    float m[8], l[8];
#pragma unroll
    for (int r = 0; r < 8; ++r) { m[r] = -INFINITY; l[r] = 0.0f; }

    const float scale = 0.17677669529663687f;  // 1/sqrt(32)

    for (int kt = 0; kt < 32; ++kt) {
        const int jb = kt * 32;

        // K^T B-operands: lane holds key column, dk is the K dim.
        const _Float16* kr0 = Kb + (size_t)(jb + nlo) * DK_ + sel * 16;
        const _Float16* kr1 = Kb + (size_t)(jb + 16 + nlo) * DK_ + sel * 16;
        v16h Bk0 = *reinterpret_cast<const v16h*>(kr0);
        v16h Bk1 = *reinterpret_cast<const v16h*>(kr1);

        v8f s0 = wmma_f16(Qa, Bk0, zero8());
        v8f s1 = wmma_f16(Qa, Bk1, zero8());

        float p0[8], p1[8], alpha[8];
#pragma unroll
        for (int r = 0; r < 8; ++r) {
            const float a0 = s0[r] * scale;
            const float a1 = s1[r] * scale;
            // Row max across the 16-lane half (rows split lanes 0-15 / 16-31)
            float t = fmaxf(a0, a1);
            t = fmaxf(t, __shfl_xor(t, 1));
            t = fmaxf(t, __shfl_xor(t, 2));
            t = fmaxf(t, __shfl_xor(t, 4));
            t = fmaxf(t, __shfl_xor(t, 8));
            const float mn = fmaxf(m[r], t);
            alpha[r] = __expf(m[r] - mn);
            p0[r] = __expf(a0 - mn);
            p1[r] = __expf(a1 - mn);
            float rs = p0[r] + p1[r];
            rs += __shfl_xor(rs, 1);
            rs += __shfl_xor(rs, 2);
            rs += __shfl_xor(rs, 4);
            rs += __shfl_xor(rs, 8);
            l[r] = l[r] * alpha[r] + rs;
            m[r] = mn;
            O0[r] = O0[r] * alpha[r];
            O1[r] = O1[r] * alpha[r];
        }

        // Re-layout P from C-format to A-format via LDS (f16).
#pragma unroll
        for (int r = 0; r < 8; ++r) {
            pbuf[(r + 8 * sel) * 32 + nlo]      = (_Float16)p0[r];
            pbuf[(r + 8 * sel) * 32 + 16 + nlo] = (_Float16)p1[r];
        }
        __syncthreads();
        const _Float16* prow = pbuf + nlo * 32 + sel * 16;
        v16h Pa = combine(*reinterpret_cast<const v8h*>(prow),
                          *reinterpret_cast<const v8h*>(prow + 8));
        __syncthreads();

        // V B-operands from transposed vT: contiguous over keys.
        const _Float16* vr0 = Vb + (size_t)nlo * S_ + jb + sel * 16;
        const _Float16* vr1 = Vb + (size_t)(16 + nlo) * S_ + jb + sel * 16;
        v16h Bv0 = *reinterpret_cast<const v16h*>(vr0);
        v16h Bv1 = *reinterpret_cast<const v16h*>(vr1);

        O0 = wmma_f16(Pa, Bv0, O0);
        O1 = wmma_f16(Pa, Bv1, O1);
    }

    // Epilogue: O/l, scatter into ctx [b][p][s][h*32+dk] (f16).
    const int b = bhp >> 6;
    const int h = (bhp >> 3) & 7;
    const int p = bhp & 7;
    const size_t tokbase = ((size_t)(b * P_ + p)) * S_ + m0;
#pragma unroll
    for (int r = 0; r < 8; ++r) {
        const float inv = 1.0f / l[r];
        const size_t trow = tokbase + r + 8 * sel;
        ctx[trow * C_ + h * 32 + nlo]      = (_Float16)(O0[r] * inv);
        ctx[trow * C_ + h * 32 + 16 + nlo] = (_Float16)(O1[r] * inv);
    }
}

// ---------------------------------------------------------------------------
// Kernel 3: output projection. out = ctx(f16) @ Wo^T + bo, f32 output.
// ---------------------------------------------------------------------------
__global__ __launch_bounds__(128) void outproj(
    const _Float16* __restrict__ ctx, const _Float16* __restrict__ Woh,
    const float* __restrict__ bo, float* __restrict__ out)
{
    const int tbase = blockIdx.x * 16;
    const int lane  = threadIdx.x & 31;
    const int wv    = threadIdx.x >> 5;
    const int nlo   = lane & 15;
    const int sel   = lane >> 4;

    v8f acc[4];
#pragma unroll
    for (int i = 0; i < 4; ++i) acc[i] = zero8();

    const _Float16* xrow = ctx + (size_t)(tbase + nlo) * C_;
#pragma unroll
    for (int kk = 0; kk < 8; ++kk) {
        const _Float16* ap = xrow + kk * 32 + sel * 16;
        v16h A = combine(*reinterpret_cast<const v8h*>(ap),
                         *reinterpret_cast<const v8h*>(ap + 8));
#pragma unroll
        for (int nt = 0; nt < 4; ++nt) {
            const int obase = (wv * 4 + nt) * 16;
            const _Float16* wrow = Woh + (size_t)(obase + nlo) * C_ + kk * 32 + sel * 16;
            v16h Bv = *reinterpret_cast<const v16h*>(wrow);
            acc[nt] = wmma_f16(A, Bv, acc[nt]);
        }
    }

#pragma unroll
    for (int nt = 0; nt < 4; ++nt) {
        const int o  = (wv * 4 + nt) * 16 + nlo;
        const float bb = bo[o];
#pragma unroll
        for (int r = 0; r < 8; ++r) {
            out[(size_t)(tbase + r + 8 * sel) * C_ + o] = acc[nt][r] + bb;
        }
    }
}

// ---------------------------------------------------------------------------
extern "C" void kernel_launch(void* const* d_in, const int* in_sizes, int n_in,
                              void* d_out, int out_size, void* d_ws, size_t ws_size,
                              hipStream_t stream) {
    const float* q  = (const float*)d_in[0];
    const float* k  = (const float*)d_in[1];
    const float* v  = (const float*)d_in[2];
    const float* Wq = (const float*)d_in[3];
    const float* bq = (const float*)d_in[4];
    const float* Wk = (const float*)d_in[5];
    const float* bk = (const float*)d_in[6];
    const float* Wv = (const float*)d_in[7];
    const float* bv = (const float*)d_in[8];
    const float* Wo = (const float*)d_in[9];
    const float* bo = (const float*)d_in[10];

    _Float16* ws = (_Float16*)d_ws;
    const size_t per = (size_t)BHP_ * S_ * DK_;   // 4M halves = 8 MB each
    _Float16* qh  = ws;
    _Float16* kh  = ws + per;
    _Float16* vT  = ws + 2 * per;
    _Float16* ctx = ws + 3 * per;                  // [T_][C_] f16
    _Float16* Wqh = ws + 4 * per;                  // 4 x 64K halves = 512 KB
    _Float16* Wkh = Wqh + (size_t)C_ * C_;
    _Float16* Wvh = Wkh + (size_t)C_ * C_;
    _Float16* Woh = Wvh + (size_t)C_ * C_;

    // 65536 elements per matrix / (256 threads * 8 elem) = 32 blocks.
    cvt_weights<<<dim3((C_ * C_) / (256 * 8), 4), 256, 0, stream>>>(
        Wq, Wk, Wv, Wo, Wqh, Wkh, Wvh, Woh);
    proj_qkv<<<dim3(T_ / 16, 3), 128, 0, stream>>>(
        q, k, v, Wqh, bq, Wkh, bk, Wvh, bv, qh, kh, vT);
    attn<<<dim3(S_ / 16, BHP_), 32, 0, stream>>>(qh, kh, vT, ctx);
    outproj<<<dim3(T_ / 16), 128, 0, stream>>>(ctx, Woh, bo, (float*)d_out);
}
